// ElmanRNN_89146341195889
// MI455X (gfx1250) — matrix operand order repaired and checked by reading, not compiled, not run on hardware
//
#include <hip/hip_runtime.h>
#include <cmath>

// ---------------------------------------------------------------------------
// Elman RNN (2 layers, B=32, S=512, D=1024) for gfx1250 (MI455X).
//   Phase 1: convert+transpose weights to bf16        (parallel, one-time)
//   Phase 2: Z0[b,t,:] = x_t @ Wh0 + b0  -> d_out     (big WMMA GEMM, parallel)
//   Phase 3: persistent recurrent kernel, 512 steps, grid-sync per layer,
//            v_wmma_f32_16x16x32_bf16 for h@U and h@W GEMMs, tanh epilogue.
// ---------------------------------------------------------------------------

typedef __bf16 bf16_t;
typedef bf16_t v16bf __attribute__((ext_vector_type(16)));
typedef bf16_t v8bf  __attribute__((ext_vector_type(8)));
typedef float  v8f   __attribute__((ext_vector_type(8)));

#define DD   1024   // embedding dim
#define BB   32     // batch
#define SS   512    // sequence length
#define NBLK 16     // persistent blocks (grid barrier member count)

// ---- WMMA helper: D = A(16x32 bf16) * B(32x16 bf16) + C(16x16 f32) --------
__device__ __forceinline__ v8f wmma_bf16(v16bf a, v16bf b, v8f c) {
  return __builtin_amdgcn_wmma_f32_16x16x32_bf16(
      /*neg_a=*/false, a, /*neg_b=*/false, b,
      /*c_mod=*/(short)0, c, /*reuse_a=*/false, /*reuse_b=*/false);
}

// A-fragment (16x32, 16-bit) from a row-major bf16 matrix.
// Caller passes p = &M[row][k0] already adjusted per-lane:
//   lane<16 : row=lane,   p -> k0      : elems 0-7 = K[k0..k0+7],  8-15 = K[k0+16..23]
//   lane>=16: row=lane-16, p -> k0+8   : elems 0-7 = K[k0+8..15],  8-15 = K[k0+24..31]
__device__ __forceinline__ v16bf load_a_bf(const bf16_t* p) {
  v8bf lo = *(const v8bf*)(p);
  v8bf hi = *(const v8bf*)(p + 16);
  return __builtin_shufflevector(lo, hi, 0,1,2,3,4,5,6,7,8,9,10,11,12,13,14,15);
}

// Same A-fragment but sourced from fp32 with on-the-fly bf16 conversion.
__device__ __forceinline__ v16bf load_a_f32(const float* p) {
  v8f lo = *(const v8f*)(p);
  v8f hi = *(const v8f*)(p + 16);
  v16bf a;
#pragma unroll
  for (int i = 0; i < 8; ++i) { a[i] = (bf16_t)lo[i]; a[8 + i] = (bf16_t)hi[i]; }
  return a;
}

// ---- grid-wide barrier (agent scope, release/acquire) ----------------------
__device__ __forceinline__ void grid_barrier(unsigned* cnt, unsigned* gen,
                                             unsigned nblk) {
  __syncthreads();
  if (threadIdx.x == 0) {
    __builtin_amdgcn_fence(__ATOMIC_RELEASE, "agent");
    unsigned g = __hip_atomic_load(gen, __ATOMIC_RELAXED, __HIP_MEMORY_SCOPE_AGENT);
    unsigned a = __hip_atomic_fetch_add(cnt, 1u, __ATOMIC_ACQ_REL,
                                        __HIP_MEMORY_SCOPE_AGENT);
    if (a == nblk - 1u) {
      __hip_atomic_store(cnt, 0u, __ATOMIC_RELAXED, __HIP_MEMORY_SCOPE_AGENT);
      __hip_atomic_fetch_add(gen, 1u, __ATOMIC_ACQ_REL, __HIP_MEMORY_SCOPE_AGENT);
    } else {
      while (__hip_atomic_load(gen, __ATOMIC_ACQUIRE,
                               __HIP_MEMORY_SCOPE_AGENT) == g) {
        __builtin_amdgcn_s_sleep(2);
      }
    }
    __builtin_amdgcn_fence(__ATOMIC_ACQUIRE, "agent");
  }
  __syncthreads();
}

// ---- Phase 1: fp32 [K][N] -> bf16 transposed [N][K], LDS tile transpose ----
// grid (32,32,4), block (32,8).  m: 0=Wh0 1=Uh0 2=Wh1 3=Uh1
__global__ void prep_weights(const float* __restrict__ Wh,
                             const float* __restrict__ Uh,
                             bf16_t* __restrict__ wt) {
  __shared__ float tile[32][33];
  int m = blockIdx.z;
  const float* src = (m == 0) ? Wh
                   : (m == 1) ? Uh
                   : (m == 2) ? (Wh + (size_t)DD * DD)
                              : (Uh + (size_t)DD * DD);
  bf16_t* dst = wt + (size_t)m * DD * DD;
  int k0 = blockIdx.x * 32, n0 = blockIdx.y * 32;
#pragma unroll
  for (int i = threadIdx.y; i < 32; i += 8)
    tile[i][threadIdx.x] = src[(size_t)(k0 + i) * DD + n0 + threadIdx.x];
  __syncthreads();
#pragma unroll
  for (int i = threadIdx.y; i < 32; i += 8)
    dst[(size_t)(n0 + i) * DD + k0 + threadIdx.x] = (bf16_t)tile[threadIdx.x][i];
}

// ---- init: zero h state (double-buffered), reset barrier words -------------
__global__ void init_state(bf16_t* __restrict__ h, unsigned* __restrict__ sync,
                           int nelem) {
  int i = blockIdx.x * blockDim.x + threadIdx.x;
  if (i < nelem) h[i] = (bf16_t)0.0f;
  if (i < 8) sync[i] = 0u;
}

// ---- Phase 2: Z0 = X @ Wh0 + b0 (written into d_out, [b][t][d]) ------------
// grid (512,16), block 128 (4 waves). Each wave: one 16-col tile, both
// batch halves, K-loop 32 x wmma_bf16 per accumulator.
__global__ void z0_gemm(const float* __restrict__ X,
                        const bf16_t* __restrict__ w0t,
                        const float* __restrict__ bias,
                        float* __restrict__ out) {
  int t    = blockIdx.x;
  int wave = threadIdx.x >> 5;
  int lane = threadIdx.x & 31;
  int ln   = lane & 15;
  int hi   = lane >> 4;
  int n    = (blockIdx.y * 4 + wave) * 16 + ln;   // global column

  float b0 = bias[n];
  v8f c0, c1;
#pragma unroll
  for (int r = 0; r < 8; ++r) { c0[r] = b0; c1[r] = b0; }

  const float*  xr0  = X + ((size_t)ln * SS + t) * DD + hi * 8;        // b = ln
  const float*  xr1  = X + ((size_t)(16 + ln) * SS + t) * DD + hi * 8; // b = 16+ln
  const bf16_t* wrow = w0t + (size_t)n * DD + hi * 8;

#pragma unroll 4
  for (int k0 = 0; k0 < DD; k0 += 32) {
    v16bf a0 = load_a_f32(xr0 + k0);
    v16bf a1 = load_a_f32(xr1 + k0);
    v16bf bm = load_a_bf(wrow + k0);
    c0 = wmma_bf16(a0, bm, c0);
    c1 = wmma_bf16(a1, bm, c1);
  }
#pragma unroll
  for (int r = 0; r < 8; ++r) {
    int m = r + hi * 8;
    out[((size_t)m * SS + t) * DD + n]        = c0[r];
    out[((size_t)(16 + m) * SS + t) * DD + n] = c1[r];
  }
}

// ---- Phase 3: persistent recurrence. 16 blocks x 128 threads = 64 waves ----
// Wave g owns columns [16g,16g+16). Per step:
//   L0: acc = Z0(from d_out) + h0_prev @ U0 ; tanh -> h0_cur   ; barrier
//   L1: acc = b1 + h0_cur @ W1 + h1_prev @ U1 ; tanh -> out, h1 ; barrier
__global__ void rnn_recurrent(const float* __restrict__ bias,
                              const bf16_t* __restrict__ wt,
                              bf16_t* __restrict__ h0,
                              bf16_t* __restrict__ h1,
                              float* __restrict__ out,
                              unsigned* __restrict__ cnt,
                              unsigned* __restrict__ gen) {
  const bf16_t* u0t = wt + (size_t)1 * DD * DD;
  const bf16_t* w1t = wt + (size_t)2 * DD * DD;
  const bf16_t* u1t = wt + (size_t)3 * DD * DD;

  int wave = threadIdx.x >> 5;
  int lane = threadIdx.x & 31;
  int ln   = lane & 15;
  int hi   = lane >> 4;
  int n    = (blockIdx.x * 4 + wave) * 16 + ln;   // owned column

  float b1 = bias[DD + n];
  const bf16_t* u0row = u0t + (size_t)n * DD + hi * 8;
  const bf16_t* w1row = w1t + (size_t)n * DD + hi * 8;
  const bf16_t* u1row = u1t + (size_t)n * DD + hi * 8;

  for (int t = 0; t < SS; ++t) {
    int p = t & 1, q = p ^ 1;
    const bf16_t* h0prev = h0 + (size_t)q * BB * DD;
    bf16_t*       h0cur  = h0 + (size_t)p * BB * DD;
    const bf16_t* h1prev = h1 + (size_t)q * BB * DD;
    bf16_t*       h1cur  = h1 + (size_t)p * BB * DD;

    // ---------- layer 0 ----------
    v8f c0, c1;
#pragma unroll
    for (int r = 0; r < 8; ++r) {          // preload Z0 (x@Wh0 + b0)
      int m = r + hi * 8;
      c0[r] = out[((size_t)m * SS + t) * DD + n];
      c1[r] = out[((size_t)(16 + m) * SS + t) * DD + n];
    }
    {
      const bf16_t* a0p = h0prev + (size_t)ln * DD + hi * 8;
      const bf16_t* a1p = h0prev + (size_t)(16 + ln) * DD + hi * 8;
#pragma unroll 4
      for (int k0 = 0; k0 < DD; k0 += 32) {
        __builtin_prefetch(u0row + k0 + 256, 0, 0);
        v16bf a0 = load_a_bf(a0p + k0);
        v16bf a1 = load_a_bf(a1p + k0);
        v16bf bm = load_a_bf(u0row + k0);
        c0 = wmma_bf16(a0, bm, c0);
        c1 = wmma_bf16(a1, bm, c1);
      }
    }
#pragma unroll
    for (int r = 0; r < 8; ++r) {          // tanh + broadcast h0_t
      int m = r + hi * 8;
      h0cur[(size_t)m * DD + n]        = (bf16_t)tanhf(c0[r]);
      h0cur[(size_t)(16 + m) * DD + n] = (bf16_t)tanhf(c1[r]);
    }
    grid_barrier(cnt, gen, NBLK);

    // ---------- layer 1 ----------
#pragma unroll
    for (int r = 0; r < 8; ++r) { c0[r] = b1; c1[r] = b1; }
    {
      const bf16_t* x0 = h0cur + (size_t)ln * DD + hi * 8;
      const bf16_t* x1 = h0cur + (size_t)(16 + ln) * DD + hi * 8;
      const bf16_t* y0 = h1prev + (size_t)ln * DD + hi * 8;
      const bf16_t* y1 = h1prev + (size_t)(16 + ln) * DD + hi * 8;
#pragma unroll 2
      for (int k0 = 0; k0 < DD; k0 += 32) {
        __builtin_prefetch(w1row + k0 + 256, 0, 0);
        __builtin_prefetch(u1row + k0 + 256, 0, 0);
        v16bf a0 = load_a_bf(x0 + k0);
        v16bf a1 = load_a_bf(x1 + k0);
        v16bf bw = load_a_bf(w1row + k0);
        c0 = wmma_bf16(a0, bw, c0);
        c1 = wmma_bf16(a1, bw, c1);
        v16bf a2 = load_a_bf(y0 + k0);
        v16bf a3 = load_a_bf(y1 + k0);
        v16bf bu = load_a_bf(u1row + k0);
        c0 = wmma_bf16(a2, bu, c0);
        c1 = wmma_bf16(a3, bu, c1);
      }
    }
#pragma unroll
    for (int r = 0; r < 8; ++r) {          // tanh -> final output + h1 state
      int m = r + hi * 8;
      float v0 = tanhf(c0[r]);
      float v1 = tanhf(c1[r]);
      out[((size_t)m * SS + t) * DD + n]        = v0;
      out[((size_t)(16 + m) * SS + t) * DD + n] = v1;
      h1cur[(size_t)m * DD + n]        = (bf16_t)v0;
      h1cur[(size_t)(16 + m) * DD + n] = (bf16_t)v1;
    }
    grid_barrier(cnt, gen, NBLK);
  }
}

// ---------------------------------------------------------------------------
extern "C" void kernel_launch(void* const* d_in, const int* in_sizes, int n_in,
                              void* d_out, int out_size, void* d_ws, size_t ws_size,
                              hipStream_t stream) {
  const float* X  = (const float*)d_in[0];   // [32,512,1024]
  const float* Wh = (const float*)d_in[1];   // [2,1024,1024]
  const float* Uh = (const float*)d_in[2];   // [2,1024,1024]
  const float* bb = (const float*)d_in[3];   // [2,1024]
  float* out = (float*)d_out;                // [32,512,1024]

  char* ws = (char*)d_ws;
  const size_t WT_BYTES = (size_t)4 * DD * DD * sizeof(bf16_t);     // 8 MB
  const size_t H_BYTES  = (size_t)2 * BB * DD * sizeof(bf16_t);     // 128 KB each
  bf16_t*   wt   = (bf16_t*)ws;
  bf16_t*   h0   = (bf16_t*)(ws + WT_BYTES);
  bf16_t*   h1   = (bf16_t*)(ws + WT_BYTES + H_BYTES);
  unsigned* sync = (unsigned*)(ws + WT_BYTES + 2 * H_BYTES);

  // 1) weights -> bf16 transposed [N][K]
  prep_weights<<<dim3(32, 32, 4), dim3(32, 8), 0, stream>>>(Wh, Uh, wt);

  // 2) zero h state (ws is poisoned) + reset barrier counters
  int hElems = (int)(2 * 2 * BB * DD);
  init_state<<<(hElems + 255) / 256, 256, 0, stream>>>(h0, sync, hElems);

  // 3) Z0 = X @ Wh0 + b0 into d_out (massively parallel WMMA GEMM)
  z0_gemm<<<dim3(SS, 16), 128, 0, stream>>>(X, wt, bb, out);

  // 4) persistent recurrence, 512 steps, grid-synced
  rnn_recurrent<<<NBLK, 128, 0, stream>>>(bb, wt, h0, h1, out, sync, sync + 1);
}
